// Proto_decoding_19155554140238
// MI455X (gfx1250) — matrix operand order repaired and checked by reading, not compiled, not run on hardware
//
#include <hip/hip_runtime.h>

// ---------------------------------------------------------------------------
// Proto decoding pipeline for MI455X (gfx1250, wave32, WMMA bf16)
// ---------------------------------------------------------------------------
#define NPIX   131072      // B*H*W = 4*128*256
#define CDIM   256
#define HH     128
#define WW     256
#define KCLS   19
#define MPROTO 10
#define KM     190
#define KCONV  2304        // 9 * 256

typedef __bf16 bf16_t;
typedef __attribute__((ext_vector_type(16))) __bf16 v16bf;
typedef __attribute__((ext_vector_type(8)))  float  v8f;
typedef __attribute__((ext_vector_type(4)))  int    i4v;

union FragAB { v16bf v; i4v i[2]; };

__device__ __forceinline__ bf16_t f2bf(float f) {
  unsigned u = __builtin_bit_cast(unsigned, f);
  unsigned r = u + 0x7FFFu + ((u >> 16) & 1u);      // round-to-nearest-even
  unsigned short h = (unsigned short)(r >> 16);
  return __builtin_bit_cast(bf16_t, h);
}
__device__ __forceinline__ float bf2f(bf16_t b) {
  unsigned short h = __builtin_bit_cast(unsigned short, b);
  unsigned u = ((unsigned)h) << 16;
  return __builtin_bit_cast(float, u);
}
__device__ __forceinline__ float wave_sum(float v) {
#pragma unroll
  for (int o = 16; o > 0; o >>= 1) v += __shfl_xor(v, o, 32);
  return v;
}
// 256-thread block reduction (8 wave32's); sh must hold 9 floats.
__device__ __forceinline__ float blk_sum256(float v, float* sh) {
  v = wave_sum(v);
  if ((threadIdx.x & 31) == 0) sh[threadIdx.x >> 5] = v;
  __syncthreads();
  if (threadIdx.x == 0) { float t = 0.f; for (int i = 0; i < 8; i++) t += sh[i]; sh[8] = t; }
  __syncthreads();
  float r = sh[8];
  __syncthreads();
  return r;
}

// ------------------------- prep kernels ------------------------------------
__global__ void k_proto_norm(const float* __restrict__ P, float* __restrict__ Pn) {
  __shared__ float sh[9];
  int row = blockIdx.x, c = threadIdx.x;
  float v = P[row * CDIM + c];
  float ss = blk_sum256(v * v, sh);
  float nrm = fmaxf(sqrtf(ss), 1e-12f);
  Pn[row * CDIM + c] = v / nrm;
}

// Build fragment-ready B matrices: dst[((kb*NT+nt)*32+lane)*16+e] = B[k][n] bf16
// B layout per ISA: lane holds col n = nt*16+(lane&15), K = kb*32 + (lane>>4)*8 + (e&7) + (e>>3)*16.
// mode 0: conv  B[k][n] = conv_w[n][cin][dy][dx], k=(dy*3+dx)*256+cin
// mode 1: B[k][n] = src[n*256+k] (1x1 weights transposed; also prototypes)
__global__ void k_build_frag(const float* __restrict__ src, bf16_t* __restrict__ dst,
                             int mode, int KB, int NT, int Krows, int Ncols) {
  int idx = blockIdx.x * 256 + threadIdx.x;
  int tot = KB * NT * 512;
  if (idx >= tot) return;
  int e = idx & 15, lane = (idx >> 4) & 31, fr = idx >> 9;
  int nt = fr % NT, kb = fr / NT;
  int k = kb * 32 + ((lane >> 4) * 8) + (e & 7) + ((e >> 3) * 16);
  int n = nt * 16 + (lane & 15);
  float v = 0.f;
  if (k < Krows && n < Ncols) {
    if (mode == 0) {
      int fidx = k >> 8, cin = k & 255;
      int dy = fidx / 3, dx = fidx - 3 * dy;
      v = src[((n * CDIM + cin) * 3 + dy) * 3 + dx];
    } else {
      v = src[n * CDIM + k];
    }
  }
  dst[idx] = f2bf(v);
}

// x NCHW f32  ->  xT [pixel][c] bf16 (NHWC)
__global__ void k_prep_xT(const float* __restrict__ x, bf16_t* __restrict__ xT) {
  size_t idx = (size_t)blockIdx.x * 256 + threadIdx.x;    // N*C threads
  int n = (int)(idx >> 8), c = (int)(idx & 255);
  int w = n & 255, h = (n >> 8) & 127, b = n >> 15;
  xT[idx] = f2bf(x[(((size_t)b * CDIM + c) * HH + h) * WW + w]);
}

// ------------------------- GEMM kernels ------------------------------------
// conv3x3 implicit GEMM: per wave: one 16-pixel M-tile x 4 N-tiles, K=2304.
// Out-of-bounds taps read a 512B zero page via pointer select (no exec masks);
// all inner loads are base+immediate so 8 k-steps pipeline freely.
__global__ __launch_bounds__(128) void k_conv_gemm(
    const bf16_t* __restrict__ xT, const bf16_t* __restrict__ Wc,
    const bf16_t* __restrict__ zpage,
    const float* __restrict__ cb, const float* __restrict__ s1,
    const float* __restrict__ b1, const float* __restrict__ m1,
    const float* __restrict__ v1, bf16_t* __restrict__ y) {
  int lane = threadIdx.x & 31, wid = threadIdx.x >> 5;
  int mt = blockIdx.x * 4 + wid;
  int nq = blockIdx.y;                       // n-tiles nq*4 .. nq*4+3
  int m0 = mt << 4;
  int p = m0 + (lane & 15);
  int pw = p & 255, ph = (p >> 8) & 127, pbt = p >> 15;
  int g0 = (lane >> 4) << 3;
  v8f acc[4];
#pragma unroll
  for (int t = 0; t < 4; t++)
#pragma unroll
    for (int r = 0; r < 8; r++) acc[t][r] = 0.f;

  const bf16_t* zp = zpage + g0;
  const bf16_t* Bp = Wc + (size_t)(nq * 4) * 512 + (size_t)lane * 16;
  __builtin_prefetch((const void*)Bp, 0, 1);

  for (int dy = 0; dy < 3; dy++) {
    int hh2 = ph + dy - 1;
    bool rowok = (hh2 >= 0) & (hh2 < HH);
    const bf16_t* rowbase = xT + ((size_t)(pbt * HH + hh2) * WW) * CDIM + g0;
    for (int dx = 0; dx < 3; dx++) {
      int ww2 = pw + dx - 1;
      bool ok = rowok & (ww2 >= 0) & (ww2 < WW);
      const bf16_t* ap = ok ? (rowbase + (size_t)ww2 * CDIM) : zp;
#pragma unroll
      for (int c8 = 0; c8 < 8; c8++) {
        FragAB fa;
        fa.i[0] = *(const i4v*)(ap + c8 * 32);
        fa.i[1] = *(const i4v*)(ap + c8 * 32 + 16);
#pragma unroll
        for (int t = 0; t < 4; t++) {
          FragAB fb;
          const bf16_t* fp = Bp + (size_t)(c8 * 16 + t) * 512;
          fb.i[0] = *(const i4v*)fp;
          fb.i[1] = *(const i4v*)(fp + 8);
          acc[t] = __builtin_amdgcn_wmma_f32_16x16x32_bf16(
              false, fa.v, false, fb.v, (short)0, acc[t], false, false);
        }
      }
      Bp += 8 * 16 * 512;                    // next (dy,dx) weight block
    }
  }
  int ro = (lane >> 4) << 3;
#pragma unroll
  for (int t = 0; t < 4; t++) {
    int c = (nq * 4 + t) * 16 + (lane & 15);
    float al = s1[c] * rsqrtf(v1[c] + 1e-5f);
    float be = (cb[c] - m1[c]) * al + b1[c];
#pragma unroll
    for (int r = 0; r < 8; r++) {
      int pm = m0 + ro + r;
      float val = fmaxf(acc[t][r] * al + be, 0.f);
      y[(size_t)pm * CDIM + c] = f2bf(val);
    }
  }
}

// 1x1 conv (pa) + BN + ReLU, K=256
__global__ __launch_bounds__(128) void k_gemm_pa(
    const bf16_t* __restrict__ A, const bf16_t* __restrict__ Wf,
    const float* __restrict__ bias, const float* __restrict__ s2,
    const float* __restrict__ b2, const float* __restrict__ m2,
    const float* __restrict__ v2, bf16_t* __restrict__ out) {
  int lane = threadIdx.x & 31, wid = threadIdx.x >> 5;
  int mt = blockIdx.x * 4 + wid;
  int nq = blockIdx.y;
  int m0 = mt << 4;
  int p = m0 + (lane & 15);
  int g0 = (lane >> 4) << 3;
  v8f acc[4];
#pragma unroll
  for (int t = 0; t < 4; t++)
#pragma unroll
    for (int r = 0; r < 8; r++) acc[t][r] = 0.f;

  const bf16_t* ap = A + (size_t)p * CDIM + g0;
  const bf16_t* Bp = Wf + (size_t)(nq * 4) * 512 + (size_t)lane * 16;
#pragma unroll
  for (int kb = 0; kb < 8; kb++) {
    FragAB fa;
    fa.i[0] = *(const i4v*)(ap + kb * 32);
    fa.i[1] = *(const i4v*)(ap + kb * 32 + 16);
#pragma unroll
    for (int t = 0; t < 4; t++) {
      FragAB fb;
      const bf16_t* fp = Bp + (size_t)(kb * 16 + t) * 512;
      fb.i[0] = *(const i4v*)fp;
      fb.i[1] = *(const i4v*)(fp + 8);
      acc[t] = __builtin_amdgcn_wmma_f32_16x16x32_bf16(
          false, fa.v, false, fb.v, (short)0, acc[t], false, false);
    }
  }
  int ro = (lane >> 4) << 3;
#pragma unroll
  for (int t = 0; t < 4; t++) {
    int c = (nq * 4 + t) * 16 + (lane & 15);
    float al = s2[c] * rsqrtf(v2[c] + 1e-5f);
    float be = (bias[c] - m2[c]) * al + b2[c];
#pragma unroll
    for (int r = 0; r < 8; r++) {
      int pm = m0 + ro + r;
      float val = fmaxf(acc[t][r] * al + be, 0.f);
      out[(size_t)pm * CDIM + c] = f2bf(val);
    }
  }
}

// 1x1 conv (pb) + per-pixel l2n -> layernorm -> l2n.  16 waves = full 256-wide
// row in LDS so each pixel's channel reductions are wave-parallel.
__global__ __launch_bounds__(512) void k_gemm_pb(
    const bf16_t* __restrict__ A, const bf16_t* __restrict__ Wf,
    const float* __restrict__ bias, const float* __restrict__ fg,
    const float* __restrict__ fb, bf16_t* __restrict__ cvec) {
  __shared__ float ls[16][CDIM];
  int lane = threadIdx.x & 31, wid = threadIdx.x >> 5;   // wid = n-tile 0..15
  int m0 = blockIdx.x << 4;
  int p = m0 + (lane & 15);
  int g0 = (lane >> 4) << 3;
  v8f acc;
#pragma unroll
  for (int r = 0; r < 8; r++) acc[r] = 0.f;

  const bf16_t* ap = A + (size_t)p * CDIM + g0;
  const bf16_t* Bp = Wf + (size_t)wid * 512 + (size_t)lane * 16;
#pragma unroll
  for (int kb = 0; kb < 8; kb++) {
    FragAB fa;
    fa.i[0] = *(const i4v*)(ap + kb * 32);
    fa.i[1] = *(const i4v*)(ap + kb * 32 + 16);
    FragAB fbf;
    const bf16_t* fp = Bp + (size_t)kb * 16 * 512;
    fbf.i[0] = *(const i4v*)fp;
    fbf.i[1] = *(const i4v*)(fp + 8);
    acc = __builtin_amdgcn_wmma_f32_16x16x32_bf16(
        false, fa.v, false, fbf.v, (short)0, acc, false, false);
  }
  int col = wid * 16 + (lane & 15);
  int ro = (lane >> 4) << 3;
  float bc = bias[col];
#pragma unroll
  for (int r = 0; r < 8; r++) ls[ro + r][col] = acc[r] + bc;
  __syncthreads();

  // wave wid handles pixel (m0+wid); lane covers 8 strided channels
  int pix = m0 + wid;
  float z[8], sz = 0.f, ss = 0.f;
#pragma unroll
  for (int t = 0; t < 8; t++) {
    float zv = ls[wid][lane + 32 * t];
    z[t] = zv; sz += zv; ss += zv * zv;
  }
  sz = wave_sum(sz); ss = wave_sum(ss);
  float n1 = fmaxf(sqrtf(ss), 1e-12f);               // l2n over channels
  float mu = sz / (n1 * 256.f);                      // LN on v = z/n1
  float var = ss / (n1 * n1 * 256.f) - mu * mu;
  float inv = rsqrtf(var + 1e-5f);
  float tv[8], tt = 0.f;
#pragma unroll
  for (int t = 0; t < 8; t++) {
    int c = lane + 32 * t;
    float v = z[t] / n1;
    float tc = (v - mu) * inv * fg[c] + fb[c];
    tv[t] = tc; tt += tc * tc;
  }
  tt = wave_sum(tt);
  float n2 = fmaxf(sqrtf(tt), 1e-12f);               // final l2n
#pragma unroll
  for (int t = 0; t < 8; t++) {
    int c = lane + 32 * t;
    cvec[(size_t)pix * CDIM + c] = f2bf(tv[t] / n2);
  }
}

// masks / proto_logits: _c[N,256] @ Pn^T[256,190] (padded to 192 cols)
__global__ __launch_bounds__(128) void k_gemm_masks(
    const bf16_t* __restrict__ A, const bf16_t* __restrict__ Pf,
    float* __restrict__ PL) {
  int lane = threadIdx.x & 31, wid = threadIdx.x >> 5;
  int mt = blockIdx.x * 4 + wid;
  int m0 = mt << 4;
  int p = m0 + (lane & 15);
  int g0 = (lane >> 4) << 3;
  v8f acc[12];
#pragma unroll
  for (int t = 0; t < 12; t++)
#pragma unroll
    for (int r = 0; r < 8; r++) acc[t][r] = 0.f;

  const bf16_t* ap = A + (size_t)p * CDIM + g0;
  const bf16_t* Bp = Pf + (size_t)lane * 16;
  for (int kb = 0; kb < 8; kb++) {
    FragAB fa;
    fa.i[0] = *(const i4v*)(ap + kb * 32);
    fa.i[1] = *(const i4v*)(ap + kb * 32 + 16);
#pragma unroll
    for (int nt = 0; nt < 12; nt++) {
      FragAB fb;
      const bf16_t* fp = Bp + (size_t)nt * 512;
      fb.i[0] = *(const i4v*)fp;
      fb.i[1] = *(const i4v*)(fp + 8);
      acc[nt] = __builtin_amdgcn_wmma_f32_16x16x32_bf16(
          false, fa.v, false, fb.v, (short)0, acc[nt], false, false);
    }
    Bp += 12 * 512;
  }
  int ro = (lane >> 4) << 3;
#pragma unroll
  for (int nt = 0; nt < 12; nt++) {
    int col = nt * 16 + (lane & 15);
    if (col < KM) {
#pragma unroll
      for (int r = 0; r < 8; r++) {
        int pm = m0 + ro + r;
        PL[(size_t)pm * KM + col] = acc[nt][r];
      }
    }
  }
}

// ------------------------- post / sinkhorn ---------------------------------
__global__ void k_postmask(const float* __restrict__ PL, const int* __restrict__ gt,
                           const float* __restrict__ mg, const float* __restrict__ mb,
                           float* __restrict__ outseg, unsigned char* __restrict__ corr) {
  int n = blockIdx.x * 256 + threadIdx.x;
  const float* row = PL + (size_t)n * KM;
  float mx[KCLS];
#pragma unroll
  for (int k = 0; k < KCLS; k++) {
    float m = row[k * MPROTO];
#pragma unroll
    for (int j = 1; j < MPROTO; j++) m = fmaxf(m, row[k * MPROTO + j]);
    mx[k] = m;
  }
  float mu = 0.f;
#pragma unroll
  for (int k = 0; k < KCLS; k++) mu += mx[k];
  mu /= (float)KCLS;
  float var = 0.f;
#pragma unroll
  for (int k = 0; k < KCLS; k++) { float d = mx[k] - mu; var += d * d; }
  var /= (float)KCLS;
  float inv = rsqrtf(var + 1e-5f);
  int w = n & 255, h = (n >> 8) & 127, b = n >> 15;
  float best = -1e30f; int bi = 0;
#pragma unroll
  for (int k = 0; k < KCLS; k++) {
    float o = (mx[k] - mu) * inv * mg[k] + mb[k];
    outseg[((size_t)(b * KCLS + k)) * (HH * WW) + (h << 8) + w] = o;
    if (o > best) { best = o; bi = k; }
  }
  corr[n] = (bi == gt[n]) ? 1 : 0;
}

__global__ void k_S1(const float* __restrict__ PL, const int* __restrict__ gt,
                     float* __restrict__ Qc, float* __restrict__ rs) {
  __shared__ float lrs[KM];
  if (threadIdx.x < KM) lrs[threadIdx.x] = 0.f;
  __syncthreads();
  int n = blockIdx.x * 256 + threadIdx.x;
  int k = gt[n];
#pragma unroll
  for (int j = 0; j < MPROTO; j++) {
    float q = __expf(PL[(size_t)n * KM + k * MPROTO + j] * 20.0f);  // 1/EPS_SINK
    Qc[(size_t)n * MPROTO + j] = q;
    atomicAdd(&lrs[k * MPROTO + j], q);
  }
  __syncthreads();
  if (threadIdx.x < KM) atomicAdd(&rs[threadIdx.x], lrs[threadIdx.x]);
}

__global__ void k_siter(const int* __restrict__ gt, float* __restrict__ Qc,
                        const float* __restrict__ rin, float* __restrict__ rout,
                        int do_acc) {
  __shared__ float lrs[KM];
  if (do_acc && threadIdx.x < KM) lrs[threadIdx.x] = 0.f;
  __syncthreads();
  int n = blockIdx.x * 256 + threadIdx.x;
  int k = gt[n];
  float q[MPROTO]; float cs = 0.f;
#pragma unroll
  for (int j = 0; j < MPROTO; j++) {
    q[j] = Qc[(size_t)n * MPROTO + j] / fmaxf(rin[k * MPROTO + j], 1e-12f);
    cs += q[j];
  }
  if (cs > 0.f) {
    float ic = 1.f / fmaxf(cs, 1e-12f);
#pragma unroll
    for (int j = 0; j < MPROTO; j++) q[j] *= ic;
  } else {
#pragma unroll
    for (int j = 0; j < MPROTO; j++) q[j] = 0.f;
  }
#pragma unroll
  for (int j = 0; j < MPROTO; j++) Qc[(size_t)n * MPROTO + j] = q[j];
  if (do_acc) {
#pragma unroll
    for (int j = 0; j < MPROTO; j++) atomicAdd(&lrs[k * MPROTO + j], q[j]);
  }
  __syncthreads();
  if (do_acc && threadIdx.x < KM) atomicAdd(&rout[threadIdx.x], lrs[threadIdx.x]);
}

__global__ void k_S3(const int* __restrict__ gt, const float* __restrict__ Qc,
                     const float* __restrict__ gum, const unsigned char* __restrict__ corr,
                     float* __restrict__ ptgt, int* __restrict__ selp,
                     int* __restrict__ ncnt) {
  __shared__ int lh[KM];
  if (threadIdx.x < KM) lh[threadIdx.x] = 0;
  __syncthreads();
  int n = blockIdx.x * 256 + threadIdx.x;
  int k = gt[n];
  float best = -1e30f; int bi = 0;
  float bg = -1e30f;  int bgi = 0;
#pragma unroll
  for (int j = 0; j < MPROTO; j++) {
    float q = Qc[(size_t)n * MPROTO + j];
    if (q > best) { best = q; bi = j; }
    float u = gum[((size_t)k * NPIX + n) * MPROTO + j];
    float g = -__logf(-__logf(u + 1e-20f) + 1e-20f);
    float qg = q + g;
    if (qg > bg) { bg = qg; bgi = j; }
  }
  ptgt[n] = (float)(bi + MPROTO * k);
  int sp = corr[n] ? (k * MPROTO + bgi) : -1;
  selp[n] = sp;
  if (sp >= 0) atomicAdd(&lh[sp], 1);
  __syncthreads();
  if (threadIdx.x < KM && lh[threadIdx.x]) atomicAdd(&ncnt[threadIdx.x], lh[threadIdx.x]);
}

// f[slot][c] += sum over matching pixels of _c[n][c]
__global__ void k_S4(const int* __restrict__ selp, const bf16_t* __restrict__ cvec,
                     float* __restrict__ fac) {
  __shared__ int sh[256];
  int slot = blockIdx.x, chunk = blockIdx.y, c = threadIdx.x;
  int base = chunk * 4096;
  float acc = 0.f;
  for (int t = 0; t < 16; t++) {
    sh[c] = selp[base + t * 256 + c];
    __syncthreads();
    for (int i = 0; i < 256; i++) {
      if (sh[i] == slot)
        acc += bf2f(cvec[(size_t)(base + t * 256 + i) * CDIM + c]);
    }
    __syncthreads();
  }
  atomicAdd(&fac[(size_t)slot * CDIM + c], acc);
}

__global__ void k_S5(const float* __restrict__ fac, const int* __restrict__ ncnt,
                     const float* __restrict__ Pn, float* __restrict__ newP) {
  __shared__ float sh[9];
  int row = blockIdx.x, c = threadIdx.x;
  int k = row / MPROTO;
  float fv = fac[(size_t)row * CDIM + c];
  float ss = blk_sum256(fv * fv, sh);
  int has = 0;
#pragma unroll
  for (int j = 0; j < MPROTO; j++) has += ncnt[k * MPROTO + j];
  int cnt = ncnt[row];
  float pv = Pn[(size_t)row * CDIM + c];
  float val;
  if (cnt != 0 && has > 0) {
    float nf = fmaxf(sqrtf(ss), 1e-12f);
    val = 0.999f * pv + 0.001f * (fv / nf);
  } else {
    val = pv;
  }
  float s2 = blk_sum256(val * val, sh);
  float n2 = fmaxf(sqrtf(s2), 1e-12f);
  newP[(size_t)row * CDIM + c] = val / n2;
}

__global__ void k_zero(float* p, int n) {
  int i = blockIdx.x * 256 + threadIdx.x;
  if (i < n) p[i] = 0.f;
}

// ---------------------------------------------------------------------------
extern "C" void kernel_launch(void* const* d_in, const int* in_sizes, int n_in,
                              void* d_out, int out_size, void* d_ws, size_t ws_size,
                              hipStream_t stream) {
  (void)in_sizes; (void)n_in; (void)out_size; (void)ws_size;
  const float* x      = (const float*)d_in[0];
  const int*   gt     = (const int*)d_in[1];
  const float* gum    = (const float*)d_in[2];
  const float* conv_w = (const float*)d_in[3];
  const float* conv_b = (const float*)d_in[4];
  const float* bn1_s  = (const float*)d_in[5];
  const float* bn1_b  = (const float*)d_in[6];
  const float* bn1_m  = (const float*)d_in[7];
  const float* bn1_v  = (const float*)d_in[8];
  const float* pa_w   = (const float*)d_in[9];
  const float* pa_b   = (const float*)d_in[10];
  const float* bn2_s  = (const float*)d_in[11];
  const float* bn2_b  = (const float*)d_in[12];
  const float* bn2_m  = (const float*)d_in[13];
  const float* bn2_v  = (const float*)d_in[14];
  const float* pb_w   = (const float*)d_in[15];
  const float* pb_b   = (const float*)d_in[16];
  const float* fn_g   = (const float*)d_in[17];
  const float* fn_b   = (const float*)d_in[18];
  const float* mn_g   = (const float*)d_in[19];
  const float* mn_b   = (const float*)d_in[20];
  const float* protos = (const float*)d_in[21];

  char* wsb = (char*)d_ws;
  bf16_t* xT   = (bf16_t*)(wsb);                       // 64MB (later: z1)
  bf16_t* ybuf = (bf16_t*)(wsb + (size_t)67108864);    // 64MB (later: _c bf16)
  size_t o = 134217728;
  bf16_t* Wc = (bf16_t*)(wsb + o); o += 1179648;       // conv frag 72x16
  bf16_t* Wa = (bf16_t*)(wsb + o); o += 131072;        // pa frag 8x16
  bf16_t* Wb = (bf16_t*)(wsb + o); o += 131072;        // pb frag 8x16
  bf16_t* Pm = (bf16_t*)(wsb + o); o += 98304;         // proto frag 8x12
  float*  Pn = (float*)(wsb + o);  o += 194560;        // normalized protos
  float*  Qc = (float*)(wsb + o);  o += 5242880;       // sinkhorn [N,10]
  float*  rsA = (float*)(wsb + o); o += 1024;          // row sums ping
  float*  rsB = (float*)(wsb + o); o += 1024;          // row sums pong
  int*    ncnt = (int*)(wsb + o);  o += 1024;          // per-slot counts
  float*  fac  = (float*)(wsb + o); o += 194560;       // f accumulator
  int*    selp = (int*)(wsb + o);  o += 524288;        // selected proto / -1
  unsigned char* corr = (unsigned char*)(wsb + o); o += 131072;
  bf16_t* zpage = (bf16_t*)(wsb + o); o += 512;        // zero page for OOB taps

  float* outp   = (float*)d_out;
  float* outseg = outp;                // [4,19,128,256]
  float* PL     = outp + 2490368;      // [131072,190]
  float* ptgt   = outp + 27394048;     // [131072]
  float* newP   = outp + 27525120;     // [19,10,256]

  bf16_t* z1   = xT;    // alias: xT dead after conv
  bf16_t* cvec = ybuf;  // alias: y dead after pa

  // ---- prep ----
  k_zero<<<1, 256, 0, stream>>>((float*)zpage, 128);   // 512B zero page
  k_proto_norm<<<KM, 256, 0, stream>>>(protos, Pn);
  k_build_frag<<<2304, 256, 0, stream>>>(conv_w, Wc, 0, 72, 16, KCONV, CDIM);
  k_build_frag<<<256,  256, 0, stream>>>(pa_w,   Wa, 1, 8, 16, CDIM, CDIM);
  k_build_frag<<<256,  256, 0, stream>>>(pb_w,   Wb, 1, 8, 16, CDIM, CDIM);
  k_build_frag<<<192,  256, 0, stream>>>(Pn,     Pm, 1, 8, 12, CDIM, KM);
  k_prep_xT<<<131072, 256, 0, stream>>>(x, xT);

  // ---- WMMA GEMM chain ----
  k_conv_gemm<<<dim3(2048, 4), 128, 0, stream>>>(xT, Wc, zpage, conv_b, bn1_s,
                                                 bn1_b, bn1_m, bn1_v, ybuf);
  k_gemm_pa<<<dim3(2048, 4), 128, 0, stream>>>(ybuf, Wa, pa_b, bn2_s, bn2_b,
                                               bn2_m, bn2_v, z1);
  k_gemm_pb<<<8192, 512, 0, stream>>>(z1, Wb, pb_b, fn_g, fn_b, cvec);
  k_gemm_masks<<<2048, 128, 0, stream>>>(cvec, Pm, PL);

  // ---- seg head / pred ----
  k_postmask<<<512, 256, 0, stream>>>(PL, gt, mn_g, mn_b, outseg, corr);

  // ---- masked sinkhorn (rsA,rsB,ncnt,fac are contiguous: 49408 floats) ----
  k_zero<<<194, 256, 0, stream>>>(rsA, 49408);
  k_S1<<<512, 256, 0, stream>>>(PL, gt, Qc, rsA);
  k_siter<<<512, 256, 0, stream>>>(gt, Qc, rsA, rsB, 1);
  k_zero<<<1, 256, 0, stream>>>(rsA, 256);
  k_siter<<<512, 256, 0, stream>>>(gt, Qc, rsB, rsA, 1);
  k_siter<<<512, 256, 0, stream>>>(gt, Qc, rsA, nullptr, 0);

  // ---- gumbel argmax + prototype update ----
  k_S3<<<512, 256, 0, stream>>>(gt, Qc, gum, corr, ptgt, selp, ncnt);
  k_S4<<<dim3(KM, 32), 256, 0, stream>>>(selp, cvec, fac);
  k_S5<<<KM, 256, 0, stream>>>(fac, ncnt, Pn, newP);
}
